// SwinNMP_44143673868293
// MI455X (gfx1250) — compile-verified, matrix operand order checked by reading, not run on hardware
//
#include <hip/hip_runtime.h>
#include <hip/hip_bf16.h>

// ---------------------------------------------------------------------------
// SwinNMP block for MI455X (gfx1250).
//  - All GEMMs + attention matmuls on v_wmma_f32_16x16x32_f16 (f16 in, f32 acc)
//  - RPE einsums factored through rpi-index lookups (kb/qb/ab tables in LDS)
//  - Whole activation set (~100MB) lives in the 192MB L2; HBM touch ~= in+out
// ---------------------------------------------------------------------------

typedef __attribute__((ext_vector_type(16))) _Float16 v16h;
typedef __attribute__((ext_vector_type(8)))  float    v8f;
typedef _Float16 f16;

#define NWIN_  144      // (96/8)^2 windows
#define LWIN_  256      // 8*8*4 tokens per window
#define NTOK_  36864    // 96*96*4 tokens
#define SCALE_ 0.17677669529663689f   // 32^-0.5

// dynamic LDS budget of the attention kernel (bytes)
#define ATTN_SMEM 190656

__device__ __forceinline__ v16h ld_frag(const f16* p0, const f16* p1) {
  union { v16h v; float4 f[2]; } u;
  u.f[0] = *(const float4*)p0;
  u.f[1] = *(const float4*)p1;
  return u.v;
}

__device__ __forceinline__ v8f wmma16(v16h a, v16h b, v8f c) {
  // D = A(16x32,f16) * B(32x16,f16) + C(16x16,f32)
  return __builtin_amdgcn_wmma_f32_16x16x32_f16(false, a, false, b, (short)0, c,
                                                false, false);
}

// relative-position index into the 225-entry rpe table
__device__ __forceinline__ int relidx(int pi, int pj) {
  int dy = (pi >> 3) - (pj >> 3) + 7;
  int dx = (pi & 7) - (pj & 7) + 7;
  return dy * 15 + dx;
}

// ---------------------------------------------------------------------------
// K0: convert weights + rpe tables to f16 (q_rpe pre-scaled by SCALE)
// ---------------------------------------------------------------------------
__global__ void prep_kernel(const float* __restrict__ qkv_w,
                            const float* __restrict__ proj_w,
                            const float* __restrict__ fc1_w,
                            const float* __restrict__ fc2_w,
                            const float* __restrict__ rpe,
                            f16* wq, f16* wp, f16* w1, f16* w2,
                            f16* qr, f16* kr, f16* vr) {
  int tid  = blockIdx.x * blockDim.x + threadIdx.x;
  int nthr = gridDim.x * blockDim.x;
  for (int i = tid; i < 384 * 128; i += nthr) wq[i] = (f16)qkv_w[i];
  for (int i = tid; i < 128 * 128; i += nthr) wp[i] = (f16)proj_w[i];
  for (int i = tid; i < 512 * 128; i += nthr) w1[i] = (f16)fc1_w[i];
  for (int i = tid; i < 128 * 512; i += nthr) w2[i] = (f16)fc2_w[i];
  for (int i = tid; i < 225 * 128; i += nthr) {
    int t = i >> 7, hc = i & 127, h = hc >> 5, c = hc & 31;
    const float* row = rpe + (size_t)t * 384 + h * 96;
    qr[i] = (f16)(row[c] * SCALE_);   // q_rpe (scaled, pairs with k)
    kr[i] = (f16)(row[32 + c]);       // k_rpe (pairs with scaled q)
    vr[i] = (f16)(row[64 + c]);       // v_rpe
  }
}

// ---------------------------------------------------------------------------
// K1/K5: LayerNorm over C=128.
//  MODE 0: read x in image layout with cyclic shift, write in window order
//  MODE 1: read x2 (token order), write token order
// ---------------------------------------------------------------------------
template <int MODE>
__global__ __launch_bounds__(128) void ln_kernel(const float* __restrict__ in,
                                                 const float* __restrict__ g,
                                                 const float* __restrict__ b,
                                                 f16* __restrict__ out) {
  int t = blockIdx.x;        // output token index
  int c = threadIdx.x;       // channel
  size_t src;
  if (MODE == 0) {
    int w = t >> 8, i = t & 255;
    int wy = w / 12, wx = w % 12;
    int p = i >> 2, hyp = i & 3;
    int gy = wy * 8 + (p >> 3), gx = wx * 8 + (p & 7);
    int sy = gy + 4; if (sy >= 96) sy -= 96;   // roll(-4) source
    int sx = gx + 4; if (sx >= 96) sx -= 96;
    src = (((size_t)(sy * 96 + sx)) * 4 + hyp) * 128 + c;
  } else {
    src = (size_t)t * 128 + c;
  }
  float v = in[src];
  __shared__ float s1[128], s2[128];
  s1[c] = v; s2[c] = v * v;
  __syncthreads();
  for (int off = 64; off > 0; off >>= 1) {
    if (c < off) { s1[c] += s1[c + off]; s2[c] += s2[c + off]; }
    __syncthreads();
  }
  float mean = s1[0] * (1.0f / 128.0f);
  float var  = s2[0] * (1.0f / 128.0f) - mean * mean;
  float rs   = rsqrtf(var + 1e-5f);
  out[(size_t)t * 128 + c] = (f16)((v - mean) * rs * g[c] + b[c]);
}

// ---------------------------------------------------------------------------
// Generic WMMA GEMM: C[M,N] = A[M,K] @ W[N,K]^T  (+fused epilogue)
//  block tile 64x64, 8 waves each own 16x32 output, K stepped by 32.
//  MODE 0: QKV   -> f16 out, +bias, q part scaled
//  MODE 1: PROJ  -> f32 out, +bias, window-reverse + unshift + residual(x)
//  MODE 2: FC1   -> f16 out, +bias, exact GELU
//  MODE 3: FC2   -> f32 out, +bias, +x2 residual
// ---------------------------------------------------------------------------
template <int MODE>
__global__ __launch_bounds__(256) void gemm_kernel(
    const f16* __restrict__ A, const f16* __restrict__ W,
    const float* __restrict__ bias, int N, int K,
    f16* __restrict__ outh, float* __restrict__ outf,
    const float* __restrict__ extra) {
  __shared__ __align__(16) f16 As[64][32];
  __shared__ __align__(16) f16 Wsm[64][32];
  int m0 = blockIdx.x * 64, n0 = blockIdx.y * 64;
  int tid = threadIdx.x, lane = tid & 31, wid = tid >> 5;
  int wr = wid >> 1, wc = wid & 1;
  int lrow = lane & 15;
  int kb = (lane < 16) ? 0 : 8;          // K-phase per half-wave (ISA A/B layout)
  int tr = tid >> 2, tc = (tid & 3) * 8; // cooperative tile load coords
  v8f acc0 = {}; v8f acc1 = {};
  for (int k0 = 0; k0 < K; k0 += 32) {
    if (k0) __syncthreads();
    *(float4*)&As[tr][tc]  = *(const float4*)&A[(size_t)(m0 + tr) * K + k0 + tc];
    *(float4*)&Wsm[tr][tc] = *(const float4*)&W[(size_t)(n0 + tr) * K + k0 + tc];
    if (k0 + 32 < K) {  // hint next K tile into cache (global_prefetch_b8)
      __builtin_prefetch(&A[(size_t)(m0 + tr) * K + k0 + 32 + tc], 0, 0);
      __builtin_prefetch(&W[(size_t)(n0 + tr) * K + k0 + 32 + tc], 0, 0);
    }
    __syncthreads();
    v16h af = ld_frag(&As[wr * 16 + lrow][kb], &As[wr * 16 + lrow][kb + 16]);
    v16h b0 = ld_frag(&Wsm[wc * 32 + lrow][kb], &Wsm[wc * 32 + lrow][kb + 16]);
    v16h b1 = ld_frag(&Wsm[wc * 32 + 16 + lrow][kb],
                      &Wsm[wc * 32 + 16 + lrow][kb + 16]);
    acc0 = wmma16(af, b0, acc0);
    acc1 = wmma16(af, b1, acc1);
  }
  // epilogue: D layout — lane<16: (M=r, N=lane); lane>=16: (M=r+8, N=lane-16)
  int mbase = m0 + wr * 16 + ((lane >> 4) << 3);
  for (int r = 0; r < 8; ++r) {
    int m = mbase + r;
    for (int half = 0; half < 2; ++half) {
      float acc = half ? acc1[r] : acc0[r];
      int n = n0 + wc * 32 + half * 16 + lrow;
      float val = acc + bias[n];
      if (MODE == 0) {                 // QKV: scale q part
        if (n < 128) val *= SCALE_;
        outh[(size_t)m * N + n] = (f16)val;
      } else if (MODE == 1) {          // PROJ: reverse windows+shift, residual
        int w = m >> 8, i = m & 255;
        int wy = w / 12, wx = w % 12;
        int p = i >> 2, hyp = i & 3;
        int gy = wy * 8 + (p >> 3), gx = wx * 8 + (p & 7);
        int sy = gy + 4; if (sy >= 96) sy -= 96;
        int sx = gx + 4; if (sx >= 96) sx -= 96;
        size_t dst = (((size_t)(sy * 96 + sx)) * 4 + hyp) * 128 + n;
        outf[dst] = val + extra[dst];
      } else if (MODE == 2) {          // FC1: exact GELU
        float ge = 0.5f * val * (1.0f + erff(val * 0.70710678118654752f));
        outh[(size_t)m * N + n] = (f16)ge;
      } else {                         // FC2: final residual
        size_t dst = (size_t)m * 128 + n;
        outf[dst] = val + extra[dst];
      }
    }
  }
}

// ---------------------------------------------------------------------------
// K3: attention. One workgroup per (window, head); 8 waves; ~186KB LDS.
//  scores: q@k^T via WMMA + qb/kb RPE bias lookups + analytic shift mask
//  softmax per row (wave32 shfl_xor), then P@V via K-split WMMA partials
//  plus the attn@v_rpe term via ab[r][pj] aggregation.
// ---------------------------------------------------------------------------
__global__ __launch_bounds__(256) void attn_kernel(
    const f16* __restrict__ qkv, const f16* __restrict__ qr_t,
    const f16* __restrict__ kr_t, const f16* __restrict__ vr_t,
    f16* __restrict__ out) {
  extern __shared__ __align__(16) char smem[];
  f16*   qs  = (f16*)smem;            // [256][32] scaled q
  f16*   ks  = qs + 256 * 32;         // [256][32]
  f16*   vT  = ks + 256 * 32;         // [32][256] v transposed
  f16*   qrS = vT + 32 * 256;         // [225][32] q_rpe (scaled) head slice
  f16*   krS = qrS + 225 * 32;        // [225][32] k_rpe head slice
  f16*   vrS = krS + 225 * 32;        // [225][32] v_rpe head slice
  f16*   Ps  = vrS + 225 * 32;        // [16][256] probabilities (f16)
  float* kbS = (float*)(Ps + 16 * 256); // [64][256] k . q_rpe bias
  float* Ss  = kbS + 64 * 256;        // [16][256] scores / partials
  float* qb  = Ss + 16 * 256;         // [16][64]  q . k_rpe bias
  float* ab  = qb + 16 * 64;          // [16][64]  prob aggregated per pj

  int w = blockIdx.x, h = blockIdx.y;
  int tid = threadIdx.x, lane = tid & 31, wid = tid >> 5;
  int wy8 = (w / 12) * 8, wx8 = (w % 12) * 8;

  // region id for the shift mask (computed on un-rolled img coordinates)
  auto regtok = [&](int i) {
    int p = i >> 2;
    int gy = wy8 + (p >> 3), gx = wx8 + (p & 7);
    int rr = (gy < 88) ? 0 : ((gy < 92) ? 1 : 2);
    int cc = (gx < 88) ? 0 : ((gx < 92) ? 1 : 2);
    return rr * 3 + cc;
  };

  // --- load q/k rows and v transposed (row tid) -----------------------------
  {
    const f16* base = qkv + ((size_t)w * 256 + tid) * 384 + h * 32;
    const float4* qp = (const float4*)(base);
    const float4* kp = (const float4*)(base + 128);
    const float4* vp = (const float4*)(base + 256);
    float4* qd = (float4*)&qs[tid * 32];
    float4* kd = (float4*)&ks[tid * 32];
    qd[0] = qp[0]; qd[1] = qp[1]; qd[2] = qp[2]; qd[3] = qp[3];
    kd[0] = kp[0]; kd[1] = kp[1]; kd[2] = kp[2]; kd[3] = kp[3];
    union { float4 f4[4]; f16 hh[32]; } vv;
    vv.f4[0] = vp[0]; vv.f4[1] = vp[1]; vv.f4[2] = vp[2]; vv.f4[3] = vp[3];
    for (int c = 0; c < 32; ++c) vT[c * 256 + tid] = vv.hh[c];
  }
  // --- rpe head slices ------------------------------------------------------
  for (int idx = tid; idx < 225 * 32; idx += 256) {
    int t = idx >> 5, c = idx & 31;
    size_t g = (size_t)t * 128 + h * 32 + c;
    qrS[idx] = qr_t[g];
    krS[idx] = kr_t[g];
    vrS[idx] = vr_t[g];
  }
  __syncthreads();
  // --- kb[pi][j] = k[j] . q_rpe[rpi(pi, j/4)]  ------------------------------
  for (int e = tid; e < 64 * 256; e += 256) {
    int pi = e >> 8, j = e & 255;
    const f16* kRow = &ks[j * 32];
    const f16* qRow = &qrS[relidx(pi, j >> 2) * 32];
    float s = 0.0f;
#pragma unroll
    for (int c = 0; c < 32; ++c) s += (float)kRow[c] * (float)qRow[c];
    kbS[e] = s;
  }
  __syncthreads();

  // --- stream 16 row-blocks -------------------------------------------------
  for (int I = 0; I < 16; ++I) {
    int i0 = I * 16;
    // qb[r][pj] = q[i0+r] . k_rpe[rpi((i0+r)/4, pj)]
    for (int e = tid; e < 16 * 64; e += 256) {
      int r = e >> 6, pj = e & 63;
      const f16* qRow = &qs[(i0 + r) * 32];
      const f16* kRow = &krS[relidx((i0 + r) >> 2, pj) * 32];
      float s = 0.0f;
#pragma unroll
      for (int c = 0; c < 32; ++c) s += (float)qRow[c] * (float)kRow[c];
      qb[e] = s;
    }
    // S = q @ k^T : 16 tiles of 16x16, 2 per wave
    {
      int lrow = lane & 15, kb = (lane < 16) ? 0 : 8;
      v16h af = ld_frag(&qs[(i0 + lrow) * 32 + kb], &qs[(i0 + lrow) * 32 + kb + 16]);
#pragma unroll
      for (int tp = 0; tp < 2; ++tp) {
        int jt = wid + tp * 8;
        v16h bf = ld_frag(&ks[(jt * 16 + lrow) * 32 + kb],
                          &ks[(jt * 16 + lrow) * 32 + kb + 16]);
        v8f d = {};
        d = wmma16(af, bf, d);
#pragma unroll
        for (int r2 = 0; r2 < 8; ++r2) {
          int m = r2 + ((lane >> 4) << 3);
          Ss[m * 256 + jt * 16 + lrow] = d[r2];
        }
      }
    }
    __syncthreads();
    // bias + shift/hypothesis mask
    for (int e = tid; e < 16 * 256; e += 256) {
      int r = e >> 8, j = e & 255;
      int i = i0 + r;
      float v = Ss[e] + qb[r * 64 + (j >> 2)] + kbS[(i >> 2) * 256 + j];
      bool blocked = (regtok(i) != regtok(j)) ||
                     (((i >> 2) == (j >> 2)) && (i != j));
      Ss[e] = blocked ? -1e30f : v;
    }
    __syncthreads();
    // softmax: each wave handles 2 rows; lane-strided over 256 cols
    for (int rr = wid * 2; rr < wid * 2 + 2; ++rr) {
      float vals[8];
      float mx = -3.4e38f;
#pragma unroll
      for (int t2 = 0; t2 < 8; ++t2) {
        vals[t2] = Ss[rr * 256 + lane + t2 * 32];
        mx = fmaxf(mx, vals[t2]);
      }
      for (int o = 16; o > 0; o >>= 1) mx = fmaxf(mx, __shfl_xor(mx, o, 32));
      float sum = 0.0f;
#pragma unroll
      for (int t2 = 0; t2 < 8; ++t2) { vals[t2] = __expf(vals[t2] - mx); sum += vals[t2]; }
      for (int o = 16; o > 0; o >>= 1) sum += __shfl_xor(sum, o, 32);
      float inv = 1.0f / sum;
#pragma unroll
      for (int t2 = 0; t2 < 8; ++t2) {
        float pv = vals[t2] * inv;
        Ss[rr * 256 + lane + t2 * 32] = pv;       // f32 copy for ab
        Ps[rr * 256 + lane + t2 * 32] = (f16)pv;  // f16 copy for WMMA
      }
    }
    __syncthreads();
    // ab[r][pj] = sum of 4 consecutive probabilities
    for (int e = tid; e < 16 * 64; e += 256) {
      int r = e >> 6, pj = e & 63;
      const float* row = &Ss[r * 256 + pj * 4];
      ab[e] = row[0] + row[1] + row[2] + row[3];
    }
    __syncthreads();
    // P @ V : K-split, wave wid owns K slice [wid*32, wid*32+32)
    {
      int lrow = lane & 15, kb = (lane < 16) ? 0 : 8;
      int j0 = wid * 32;
      v16h ap = ld_frag(&Ps[lrow * 256 + j0 + kb], &Ps[lrow * 256 + j0 + kb + 16]);
      v16h b0 = ld_frag(&vT[lrow * 256 + j0 + kb], &vT[lrow * 256 + j0 + kb + 16]);
      v16h b1 = ld_frag(&vT[(16 + lrow) * 256 + j0 + kb],
                        &vT[(16 + lrow) * 256 + j0 + kb + 16]);
      v8f d0 = {}, d1 = {};
      d0 = wmma16(ap, b0, d0);
      d1 = wmma16(ap, b1, d1);
#pragma unroll
      for (int r2 = 0; r2 < 8; ++r2) {   // partials into Ss[r][wid*32 + c]
        int m = r2 + ((lane >> 4) << 3);
        Ss[m * 256 + wid * 32 + lrow]      = d0[r2];
        Ss[m * 256 + wid * 32 + 16 + lrow] = d1[r2];
      }
    }
    __syncthreads();
    // reduce partials + attn@v_rpe term, store f16 token-major
    for (int e = tid; e < 16 * 32; e += 256) {
      int r = e >> 5, c = e & 31;
      float s = 0.0f;
#pragma unroll
      for (int ww = 0; ww < 8; ++ww) s += Ss[r * 256 + ww * 32 + c];
      int i = i0 + r, pi = i >> 2;
      for (int pj = 0; pj < 64; ++pj)
        s += ab[r * 64 + pj] * (float)vrS[relidx(pi, pj) * 32 + c];
      out[((size_t)(w * 256 + i)) * 128 + h * 32 + c] = (f16)s;
    }
    __syncthreads();
  }
}

// ---------------------------------------------------------------------------
extern "C" void kernel_launch(void* const* d_in, const int* in_sizes, int n_in,
                              void* d_out, int out_size, void* d_ws, size_t ws_size,
                              hipStream_t stream) {
  const float* x      = (const float*)d_in[0];
  const float* n1g    = (const float*)d_in[1];
  const float* n1b    = (const float*)d_in[2];
  const float* qkv_w  = (const float*)d_in[3];
  const float* qkv_b  = (const float*)d_in[4];
  const float* rpe    = (const float*)d_in[5];
  const float* proj_w = (const float*)d_in[6];
  const float* proj_b = (const float*)d_in[7];
  const float* n2g    = (const float*)d_in[8];
  const float* n2b    = (const float*)d_in[9];
  const float* fc1_w  = (const float*)d_in[10];
  const float* fc1_b  = (const float*)d_in[11];
  const float* fc2_w  = (const float*)d_in[12];
  const float* fc2_b  = (const float*)d_in[13];
  float* out = (float*)d_out;

  char* base = (char*)d_ws;
  size_t o = 0;
  auto carve = [&](size_t bytes) {
    char* r = base + o;
    o = (o + bytes + 255) & ~(size_t)255;
    return r;
  };
  f16* wq    = (f16*)carve(384 * 128 * 2);
  f16* wp    = (f16*)carve(128 * 128 * 2);
  f16* w1    = (f16*)carve(512 * 128 * 2);
  f16* w2    = (f16*)carve(128 * 512 * 2);
  f16* qr    = (f16*)carve(225 * 128 * 2);
  f16* kr    = (f16*)carve(225 * 128 * 2);
  f16* vr    = (f16*)carve(225 * 128 * 2);
  f16* hwin  = (f16*)carve((size_t)NTOK_ * 128 * 2);   // |
  f16* qkvh  = (f16*)carve((size_t)NTOK_ * 384 * 2);   // | contiguous -> y1
  f16* attnh = (f16*)carve((size_t)NTOK_ * 128 * 2);
  float* x2  = (float*)carve((size_t)NTOK_ * 128 * 4);
  f16* y1    = hwin;   // reuse hwin+qkvh (dead after attention) for fc1 out
  f16* ln2h  = attnh;  // reuse attention output (dead after proj) for LN2

  prep_kernel<<<64, 256, 0, stream>>>(qkv_w, proj_w, fc1_w, fc2_w, rpe,
                                      wq, wp, w1, w2, qr, kr, vr);
  ln_kernel<0><<<NTOK_, 128, 0, stream>>>(x, n1g, n1b, hwin);
  gemm_kernel<0><<<dim3(NTOK_ / 64, 384 / 64), 256, 0, stream>>>(
      hwin, wq, qkv_b, 384, 128, qkvh, nullptr, nullptr);
  attn_kernel<<<dim3(NWIN_, 4), 256, ATTN_SMEM, stream>>>(qkvh, qr, kr, vr, attnh);
  gemm_kernel<1><<<dim3(NTOK_ / 64, 128 / 64), 256, 0, stream>>>(
      attnh, wp, proj_b, 128, 128, nullptr, x2, x);
  ln_kernel<1><<<NTOK_, 128, 0, stream>>>(x2, n2g, n2b, ln2h);
  gemm_kernel<2><<<dim3(NTOK_ / 64, 512 / 64), 256, 0, stream>>>(
      ln2h, w1, fc1_b, 512, 128, y1, nullptr, nullptr);
  gemm_kernel<3><<<dim3(NTOK_ / 64, 128 / 64), 256, 0, stream>>>(
      y1, w2, fc2_b, 128, 512, nullptr, out, x2);
}
